// UndirectedAttn_56779467653668
// MI455X (gfx1250) — matrix operand order repaired and checked by reading, not compiled, not run on hardware
//
#include <hip/hip_runtime.h>
#include <hip/hip_bf16.h>

// ---------------------------------------------------------------------------
// Problem constants (fixed by the reference)
// ---------------------------------------------------------------------------
#define BB     8
#define NN_    1024
#define DD     256
#define VOCAB_ 32000
#define NMASK  512
#define TOK_MASK 1
#define TOK_PAD  0

// GEMM tiling: block tile 64(M) x 128(N) x 32(K); 8 wave32s, each owns 32x32.
#define BM 64
#define BN 128
#define BK 32
#define LDSA_STRIDE 40   // bf16 elems per LDS tile row: 64B data + 16B pad
#define LDSB_STRIDE 40
#define ABUF_BYTES (BM * LDSA_STRIDE * 2)   // 5120
#define BBUF_BYTES (BN * LDSB_STRIDE * 2)   // 10240

typedef __attribute__((ext_vector_type(4)))  float        f32x4;
typedef __attribute__((ext_vector_type(8)))  float        f32x8;
typedef __attribute__((ext_vector_type(4)))  __bf16       bf16x4;
typedef __attribute__((ext_vector_type(8)))  __bf16       bf16x8;
typedef __attribute__((ext_vector_type(16))) __bf16       bf16x16;
typedef __attribute__((ext_vector_type(4)))  unsigned int u32x4;
typedef __attribute__((ext_vector_type(8)))  int          i32x8;
typedef __attribute__((ext_vector_type(4)))  int          i32x4;

// ---------------------------------------------------------------------------
// TDM: async-load a 2D bf16 tile [rows x 32] from a row-major tensor (leading
// dim `ld` elements) into LDS at byte offset lds_off, writing each 64B row
// followed by 16B pad (-> LDS row stride 80B = 40 bf16, matching fragments).
// D# packing per CDNA5 ISA ch.8 (group0 128b, group1 256b; groups 2/3 zero).
// This toolchain's builtin is the 6-arg form:
//   (u32x4 g0, i32x8 g1, i32x4, i32x4, i32x8, i32 cpol)
// ---------------------------------------------------------------------------
__device__ __forceinline__
void tdm_load_tile(const __bf16* gsrc, unsigned lds_off, int ld, int rows, int trows)
{
    const unsigned long long ga = (unsigned long long)(const void*)gsrc;
    u32x4 g0;
    g0[0] = 1u;                                            // count=1 (valid), user mode
    g0[1] = lds_off;                                       // lds_addr (bytes)
    g0[2] = (unsigned)(ga & 0xFFFFFFFFull);                // global_addr[31:0]
    g0[3] = (unsigned)((ga >> 32) & 0x01FFFFFFull)         // global_addr[56:32]
          | (2u << 30);                                    // type = 2 ("image")
    i32x8 g1;
    g1[0] = (1 << 16)      // data_size = 2 bytes
          | (1 << 20)      // pad_enable
          | (3 << 22)      // pad_interval: 16 DWORDs (64B) between pads
          | (3 << 25);     // pad_amount: 4 DWORDs (16B)
    g1[1] = (int)(((unsigned)ld & 0xFFFFu) << 16);                         // tensor_dim0[15:0]
    g1[2] = (int)((((unsigned)ld >> 16) & 0xFFFFu)
          | (((unsigned)trows & 0xFFFFu) << 16));                          // dim0 hi | dim1 lo
    g1[3] = (int)((((unsigned)trows >> 16) & 0xFFFFu) | (32u << 16));      // dim1 hi | tile_dim0=32
    g1[4] = rows;                                                          // tile_dim1 (tile_dim2=0)
    g1[5] = ld;                                                            // tensor_dim0_stride lo32
    g1[6] = 0;                                                             // stride hi | dim1_stride lo
    g1[7] = 0;
    const i32x4 z4 = {0, 0, 0, 0};
    const i32x8 z8 = {0, 0, 0, 0, 0, 0, 0, 0};
    __builtin_amdgcn_tensor_load_to_lds(g0, g1, z4, z4, z8, 0);
}

// ---------------------------------------------------------------------------
// Generic WMMA GEMM on bf16 operands with f32 accumulate:
//   C[M,Nc] (+)= opA(A) * opB(B);  optional bf16 mirror of C written too.
//   TA=0: A [M,Kt] row-major (lda=Kt)  -> tile loaded by TDM (contiguous)
//   TA=1: A [Kt,M]          (lda=M)   -> cooperative transposing fill
//   TB=1: B [Nc,Kt] row-major (ldb=Kt)-> tile loaded by TDM (contiguous)
//   TB=0: B [Kt,Nc]          (ldb=Nc) -> cooperative transposing fill
// Double-buffered LDS: TDM prefetches tile k+1 while waves compute tile k.
// ---------------------------------------------------------------------------
template<int TA, int TB, int ACC>
__global__ __launch_bounds__(256)
void wmma_gemm_kernel(const __bf16* __restrict__ Ag, const __bf16* __restrict__ Bg,
                      float* __restrict__ Cg, __bf16* __restrict__ Cb,
                      int M, int Nc, int Kt,
                      long batchA, long batchB, long batchC,
                      int lda, int ldb, int ldc)
{
    __shared__ __attribute__((aligned(128))) __bf16 As[2 * BM * LDSA_STRIDE];
    __shared__ __attribute__((aligned(128))) __bf16 Bs[2 * BN * LDSB_STRIDE];

    const int bz = blockIdx.z;
    const int n0 = blockIdx.x * BN;
    const int m0 = blockIdx.y * BM;
    const __bf16* A = Ag + (long)bz * batchA;
    const __bf16* B = Bg + (long)bz * batchB;
    float*        C = Cg + (long)bz * batchC;

    const int tid  = threadIdx.x;
    const int wave = tid >> 5;
    const int lane = tid & 31;
    const int h    = lane >> 4;   // half-wave select (ISA fragment layout)
    const int idx  = lane & 15;

    const int wr = (wave & 1) * 32;    // wave's row offset inside block tile
    const int wc = (wave >> 1) * 32;   // wave's col offset inside block tile

    const unsigned ldsA = (unsigned)(unsigned long long)(const void*)&As[0];
    const unsigned ldsB = (unsigned)(unsigned long long)(const void*)&Bs[0];

    constexpr int NTDM = (TA == 0 ? 1 : 0) + (TB == 1 ? 1 : 0);

    f32x8 c00 = {}, c01 = {}, c10 = {}, c11 = {};
    if (ACC) {
        #pragma unroll
        for (int v = 0; v < 8; ++v) {
            const long r0 = (long)(m0 + wr + v + 8 * h) * ldc;
            const long r1 = (long)(m0 + wr + 16 + v + 8 * h) * ldc;
            c00[v] = C[r0 + (n0 + wc + idx)];
            c01[v] = C[r0 + (n0 + wc + 16 + idx)];
            c10[v] = C[r1 + (n0 + wc + idx)];
            c11[v] = C[r1 + (n0 + wc + 16 + idx)];
        }
    }

    const int P = Kt / BK;

    if (NTDM && wave == 0) {   // prologue: async-load tile 0 into buffer 0
        if (TA == 0) tdm_load_tile(&A[(long)m0 * lda], ldsA, lda, BM, M);
        if (TB == 1) tdm_load_tile(&B[(long)n0 * ldb], ldsB, ldb, BN, Nc);
    }

    for (int p = 0; p < P; ++p) {
        const int cur = p & 1;
        const int k0  = p * BK;

        if (NTDM && wave == 0) {
            if (p + 1 < P) {   // prefetch next tile into the other buffer
                const int kn = k0 + BK;
                if (TA == 0) tdm_load_tile(&A[(long)m0 * lda + kn],
                                           ldsA + (cur ^ 1) * ABUF_BYTES, lda, BM, M);
                if (TB == 1) tdm_load_tile(&B[(long)n0 * ldb + kn],
                                           ldsB + (cur ^ 1) * BBUF_BYTES, ldb, BN, Nc);
                __builtin_amdgcn_s_wait_tensorcnt(NTDM);   // current tile's ops done
            } else {
                __builtin_amdgcn_s_wait_tensorcnt(0);
            }
        }
        if (TA == 1) {   // A^T: gather 4 strided k's per thread, ds_store_b64
            #pragma unroll
            for (int i = 0; i < (BM * BK) / (256 * 4); ++i) {
                const int q  = tid + i * 256;
                const int r  = q & 63, c4 = (q >> 6) * 4;
                const __bf16* gp = &A[(long)(k0 + c4) * lda + (m0 + r)];
                bf16x4 w;
                w[0] = gp[0]; w[1] = gp[(long)lda];
                w[2] = gp[(long)lda * 2]; w[3] = gp[(long)lda * 3];
                *(bf16x4*)&As[cur * BM * LDSA_STRIDE + r * LDSA_STRIDE + c4] = w;
            }
        }
        if (TB == 0) {   // NN B: gather 4 strided k's per thread
            #pragma unroll
            for (int i = 0; i < (BN * BK) / (256 * 4); ++i) {
                const int q  = tid + i * 256;
                const int r  = q & 127, c4 = (q >> 7) * 4;
                const __bf16* gp = &B[(long)(k0 + c4) * ldb + (n0 + r)];
                bf16x4 w;
                w[0] = gp[0]; w[1] = gp[(long)ldb];
                w[2] = gp[(long)ldb * 2]; w[3] = gp[(long)ldb * 3];
                *(bf16x4*)&Bs[cur * BN * LDSB_STRIDE + r * LDSB_STRIDE + c4] = w;
            }
        }
        __syncthreads();

        // ---- fragments (16B ds loads matching ISA VGPR layouts) ----
        union { bf16x16 v; bf16x8 p[2]; } a0, a1, b0, b1;
        const __bf16* Ab = &As[cur * BM * LDSA_STRIDE];
        const __bf16* Bb = &Bs[cur * BN * LDSB_STRIDE];
        const __bf16* ar0 = &Ab[(wr + idx) * LDSA_STRIDE];
        a0.p[0] = *(const bf16x8*)(ar0 + h * 8);
        a0.p[1] = *(const bf16x8*)(ar0 + 16 + h * 8);
        const __bf16* ar1 = &Ab[(wr + 16 + idx) * LDSA_STRIDE];
        a1.p[0] = *(const bf16x8*)(ar1 + h * 8);
        a1.p[1] = *(const bf16x8*)(ar1 + 16 + h * 8);
        const __bf16* br0 = &Bb[(wc + idx) * LDSB_STRIDE];
        b0.p[0] = *(const bf16x8*)(br0 + h * 8);
        b0.p[1] = *(const bf16x8*)(br0 + 16 + h * 8);
        const __bf16* br1 = &Bb[(wc + 16 + idx) * LDSB_STRIDE];
        b1.p[0] = *(const bf16x8*)(br1 + h * 8);
        b1.p[1] = *(const bf16x8*)(br1 + 16 + h * 8);

        c00 = __builtin_amdgcn_wmma_f32_16x16x32_bf16(false, a0.v, false, b0.v, (short)0, c00, false, false);
        c01 = __builtin_amdgcn_wmma_f32_16x16x32_bf16(false, a0.v, false, b1.v, (short)0, c01, false, false);
        c10 = __builtin_amdgcn_wmma_f32_16x16x32_bf16(false, a1.v, false, b0.v, (short)0, c10, false, false);
        c11 = __builtin_amdgcn_wmma_f32_16x16x32_bf16(false, a1.v, false, b1.v, (short)0, c11, false, false);
        __syncthreads();
    }

    #pragma unroll
    for (int v = 0; v < 8; ++v) {
        const long r0 = (long)(m0 + wr + v + 8 * h) * ldc;
        const long r1 = (long)(m0 + wr + 16 + v + 8 * h) * ldc;
        C[r0 + (n0 + wc + idx)]      = c00[v];
        C[r0 + (n0 + wc + 16 + idx)] = c01[v];
        C[r1 + (n0 + wc + idx)]      = c10[v];
        C[r1 + (n0 + wc + 16 + idx)] = c11[v];
    }
    if (Cb) {   // bf16 mirror for downstream GEMM operands
        __bf16* Mb = Cb + (long)bz * batchC;
        #pragma unroll
        for (int v = 0; v < 8; ++v) {
            const long r0 = (long)(m0 + wr + v + 8 * h) * ldc;
            const long r1 = (long)(m0 + wr + 16 + v + 8 * h) * ldc;
            Mb[r0 + (n0 + wc + idx)]      = (__bf16)c00[v];
            Mb[r0 + (n0 + wc + 16 + idx)] = (__bf16)c01[v];
            Mb[r1 + (n0 + wc + idx)]      = (__bf16)c10[v];
            Mb[r1 + (n0 + wc + 16 + idx)] = (__bf16)c11[v];
        }
    }
}

// ---------------------------------------------------------------------------
// Row softmax with pad masking; reads f32 raw scores, writes bf16 S mirror.
// ---------------------------------------------------------------------------
__global__ __launch_bounds__(256)
void softmax_kernel(const float* __restrict__ S, __bf16* __restrict__ Sb,
                    const int* __restrict__ ix)
{
    __shared__ float red[256];
    const int row = blockIdx.x;            // b*N + i
    const int b   = row >> 10;
    const int i   = row & (NN_ - 1);
    const int* ixb = ix + b * NN_;
    const bool pad_i = (ixb[i] == TOK_PAD);
    const float* Srow = S  + (long)row * NN_;
    __bf16*      Brow = Sb + (long)row * NN_;

    float vals[NN_ / 256];
    float lmax = -3.0e38f;
    #pragma unroll
    for (int t = 0; t < NN_ / 256; ++t) {
        const int j = threadIdx.x + t * 256;
        const bool pm = pad_i || (ixb[j] == TOK_PAD);
        const float x = (Srow[j] + (pm ? -9999.0f : 0.0f)) * (1.0f / 16.0f);
        vals[t] = x;
        lmax = fmaxf(lmax, x);
    }
    red[threadIdx.x] = lmax; __syncthreads();
    for (int s = 128; s > 0; s >>= 1) {
        if (threadIdx.x < s) red[threadIdx.x] = fmaxf(red[threadIdx.x], red[threadIdx.x + s]);
        __syncthreads();
    }
    const float m = red[0]; __syncthreads();

    float lsum = 0.0f;
    #pragma unroll
    for (int t = 0; t < NN_ / 256; ++t) { vals[t] = __expf(vals[t] - m); lsum += vals[t]; }
    red[threadIdx.x] = lsum; __syncthreads();
    for (int s = 128; s > 0; s >>= 1) {
        if (threadIdx.x < s) red[threadIdx.x] += red[threadIdx.x + s];
        __syncthreads();
    }
    const float inv = 1.0f / red[0];

    #pragma unroll
    for (int t = 0; t < NN_ / 256; ++t) {
        const int j = threadIdx.x + t * 256;
        const bool pm = pad_i || (ixb[j] == TOK_PAD);
        Brow[j] = pm ? (__bf16)0.0f : (__bf16)(vals[t] * inv);
    }
}

// ---------------------------------------------------------------------------
// Elementwise helpers
// ---------------------------------------------------------------------------
__global__ void zerob_kernel(__bf16* __restrict__ p, long n) {
    for (long i = blockIdx.x * (long)blockDim.x + threadIdx.x; i < n;
         i += (long)gridDim.x * blockDim.x) p[i] = (__bf16)0.0f;
}

__global__ void convb_kernel(const float* __restrict__ s, __bf16* __restrict__ d, long n) {
    for (long i = blockIdx.x * (long)blockDim.x + threadIdx.x; i < n;
         i += (long)gridDim.x * blockDim.x) d[i] = (__bf16)s[i];
}

__global__ void pe_kernel(float* __restrict__ PE) {     // [N, D]
    long gid = blockIdx.x * (long)blockDim.x + threadIdx.x;
    if (gid >= (long)NN_ * DD) return;
    const int p  = (int)(gid / DD);
    const int dd = (int)(gid % DD);
    const int half = DD / 2;
    const int k = (dd < half) ? dd : dd - half;
    const float ang = (float)p * powf(10000.0f, -(float)k / (float)half);
    PE[gid] = (dd < half) ? sinf(ang) : cosf(ang);
}

__global__ void initx_kernel(const int* __restrict__ ix, const float* __restrict__ emb,
                             float* __restrict__ X) {   // [B*N, D]
    long gid = blockIdx.x * (long)blockDim.x + threadIdx.x;
    if (gid >= (long)BB * NN_ * DD) return;
    const long bn = gid / DD;
    const int  dd = (int)(gid % DD);
    const int  t  = ix[bn];
    X[gid] = (t == TOK_MASK) ? 0.0f : emb[(long)t * DD + dd];
}

__global__ void addpe_kernel(const float* __restrict__ X, const float* __restrict__ PE,
                             __bf16* __restrict__ Xpb) {
    long gid = blockIdx.x * (long)blockDim.x + threadIdx.x;
    if (gid >= (long)BB * NN_ * DD) return;
    Xpb[gid] = (__bf16)(X[gid] + PE[gid % ((long)NN_ * DD)]);
}

__global__ void updatex_kernel(const int* __restrict__ ix, const float* __restrict__ Xm,
                               float* __restrict__ X) {
    long gid = blockIdx.x * (long)blockDim.x + threadIdx.x;
    if (gid >= (long)BB * NN_ * DD) return;
    if (ix[gid / DD] == TOK_MASK) X[gid] = Xm[gid];
}

__global__ void gather_kernel(const float* __restrict__ Xm, const int* __restrict__ midx,
                              __bf16* __restrict__ Xgb) {  // [NMASK, D] bf16
    long gid = blockIdx.x * (long)blockDim.x + threadIdx.x;
    if (gid >= (long)NMASK * DD) return;
    const int r  = (int)(gid / DD);
    const int dd = (int)(gid % DD);
    Xgb[gid] = (__bf16)Xm[(long)midx[r] * DD + dd];
}

// ---------------------------------------------------------------------------
// Host-side launch helper
// ---------------------------------------------------------------------------
static inline void gemm(hipStream_t s, const __bf16* A, const __bf16* B,
                        float* C, __bf16* Cb,
                        int M, int Nc, int Kt, long bA, long bB, long bC,
                        int lda, int ldb, int ldc, int tA, int tB, int acc, int batch)
{
    dim3 g(Nc / BN, M / BM, batch);
    dim3 t(256);
#define LAUNCH(TA, TB, AC) \
    wmma_gemm_kernel<TA, TB, AC><<<g, t, 0, s>>>(A, B, C, Cb, M, Nc, Kt, bA, bB, bC, lda, ldb, ldc)
    const int key = tA * 4 + tB * 2 + acc;
    switch (key) {
        case 0: LAUNCH(0, 0, 0); break;
        case 1: LAUNCH(0, 0, 1); break;
        case 2: LAUNCH(0, 1, 0); break;
        case 3: LAUNCH(0, 1, 1); break;
        case 4: LAUNCH(1, 0, 0); break;
        case 5: LAUNCH(1, 0, 1); break;
        case 6: LAUNCH(1, 1, 0); break;
        default: LAUNCH(1, 1, 1); break;
    }
#undef LAUNCH
}

extern "C" void kernel_launch(void* const* d_in, const int* in_sizes, int n_in,
                              void* d_out, int out_size, void* d_ws, size_t ws_size,
                              hipStream_t stream)
{
    (void)in_sizes; (void)n_in; (void)out_size; (void)ws_size;
    const int*   ix   = (const int*)d_in[0];
    const int*   midx = (const int*)d_in[1];
    const float* emb  = (const float*)d_in[2];
    const float* Wq   = (const float*)d_in[3];
    const float* Wk   = (const float*)d_in[4];
    const float* Wv   = (const float*)d_in[5];
    float*       out  = (float*)d_out;

    const long ND  = (long)NN_ * DD;        // 262144
    const long NN2 = (long)NN_ * NN_;       // 1048576
    const long BND = (long)BB * ND;         // 2097152
    const long BNN = (long)BB * NN2;        // 8388608
    const long DD2 = (long)DD * DD;         // 65536

    // f32 region
    float* X  = (float*)d_ws;
    float* Q  = X  + BND;
    float* K  = Q  + BND;
    float* V  = K  + BND;
    float* H  = V  + BND;
    float* Xm = H  + BND;
    float* S  = Xm + BND;
    float* PE = S  + BNN;
    // bf16 region (persistent operand mirrors)
    __bf16* Xpb = (__bf16*)(PE + ND);
    __bf16* Qb  = Xpb + BND;
    __bf16* Kb  = Qb  + BND;
    __bf16* Vb  = Kb  + BND;
    __bf16* Hb  = Vb  + BND;
    __bf16* Sb  = Hb  + BND;
    __bf16* Wqb = Sb  + BNN;
    __bf16* Wkb = Wqb + DD2;
    __bf16* Wvb = Wkb + DD2;
    __bf16* Eb  = Wvb + DD2;                    // emb mirror [VOCAB, D]
    __bf16* Xgb = Eb  + (long)VOCAB_ * DD;      // gathered rows [NMASK, D]

    const int TPB = 256;
    const int gBND = (int)((BND + TPB - 1) / TPB);

    // ---- one-time setup ----
    pe_kernel<<<(int)((ND + TPB - 1) / TPB), TPB, 0, stream>>>(PE);
    initx_kernel<<<gBND, TPB, 0, stream>>>(ix, emb, X);
    convb_kernel<<<256, TPB, 0, stream>>>(Wq, Wqb, DD2);
    convb_kernel<<<256, TPB, 0, stream>>>(Wk, Wkb, DD2);
    convb_kernel<<<256, TPB, 0, stream>>>(Wv, Wvb, DD2);
    convb_kernel<<<4096, TPB, 0, stream>>>(emb, Eb, (long)VOCAB_ * DD);
    zerob_kernel<<<4096, TPB, 0, stream>>>(Qb, 4 * BND + BNN);   // Qb,Kb,Vb,Hb,Sb

    for (int it = 0; it < 3; ++it) {
        // Xp = X + PE  (bf16 operand mirror only)
        addpe_kernel<<<gBND, TPB, 0, stream>>>(X, PE, Xpb);

        // Q = Xp @ Wq^T + S @ K
        gemm(stream, Xpb, Wqb, Q, nullptr, NN_, DD, DD,  ND, 0, ND,  DD, DD, DD, 0, 1, 0, BB);
        gemm(stream, Sb,  Kb,  Q, Qb,      NN_, DD, NN_, NN2, ND, ND, NN_, DD, DD, 0, 0, 1, BB);
        // K = Xp @ Wk^T + S^T @ Q
        gemm(stream, Xpb, Wkb, K, nullptr, NN_, DD, DD,  ND, 0, ND,  DD, DD, DD, 0, 1, 0, BB);
        gemm(stream, Sb,  Qb,  K, Kb,      NN_, DD, NN_, NN2, ND, ND, NN_, DD, DD, 1, 0, 1, BB);
        // V = Xp @ Wv^T + S^T @ H
        gemm(stream, Xpb, Wvb, V, nullptr, NN_, DD, DD,  ND, 0, ND,  DD, DD, DD, 0, 1, 0, BB);
        gemm(stream, Sb,  Hb,  V, Vb,      NN_, DD, NN_, NN2, ND, ND, NN_, DD, DD, 1, 0, 1, BB);

        // S = softmax((Q K^T + H V^T + mask)/sqrt(d)), zero pad rows/cols
        gemm(stream, Qb, Kb, S, nullptr, NN_, NN_, DD, ND, ND, NN2, DD, DD, NN_, 0, 1, 0, BB);
        gemm(stream, Hb, Vb, S, nullptr, NN_, NN_, DD, ND, ND, NN2, DD, DD, NN_, 0, 1, 1, BB);
        softmax_kernel<<<BB * NN_, TPB, 0, stream>>>(S, Sb, ix);

        // H = S @ V
        gemm(stream, Sb, Vb, H, Hb, NN_, DD, NN_, NN2, ND, ND, NN_, DD, DD, 0, 0, 0, BB);

        // S = update_S again (with new H)
        gemm(stream, Qb, Kb, S, nullptr, NN_, NN_, DD, ND, ND, NN2, DD, DD, NN_, 0, 1, 0, BB);
        gemm(stream, Hb, Vb, S, nullptr, NN_, NN_, DD, ND, ND, NN2, DD, DD, NN_, 0, 1, 1, BB);
        softmax_kernel<<<BB * NN_, TPB, 0, stream>>>(S, Sb, ix);

        // V = Xp @ Wv^T + S^T @ H
        gemm(stream, Xpb, Wvb, V, nullptr, NN_, DD, DD,  ND, 0, ND,  DD, DD, DD, 0, 1, 0, BB);
        gemm(stream, Sb,  Hb,  V, Vb,      NN_, DD, NN_, NN2, ND, ND, NN_, DD, DD, 1, 0, 1, BB);
        // K = Xp @ Wk^T + S^T @ Q
        gemm(stream, Xpb, Wkb, K, nullptr, NN_, DD, DD,  ND, 0, ND,  DD, DD, DD, 0, 1, 0, BB);
        gemm(stream, Sb,  Qb,  K, Kb,      NN_, DD, NN_, NN2, ND, ND, NN_, DD, DD, 1, 0, 1, BB);
        // Q = Xp @ Wq^T + S @ K   (new K)
        gemm(stream, Xpb, Wqb, Q, nullptr, NN_, DD, DD,  ND, 0, ND,  DD, DD, DD, 0, 1, 0, BB);
        gemm(stream, Sb,  Kb,  Q, Qb,      NN_, DD, NN_, NN2, ND, ND, NN_, DD, DD, 0, 0, 1, BB);

        // Xm = V@Wv + K@Wk + Q@Wq   (no transpose; f32 only)
        gemm(stream, Vb, Wvb, Xm, nullptr, NN_, DD, DD, ND, 0, ND, DD, DD, DD, 0, 0, 0, BB);
        gemm(stream, Kb, Wkb, Xm, nullptr, NN_, DD, DD, ND, 0, ND, DD, DD, DD, 0, 0, 1, BB);
        gemm(stream, Qb, Wqb, Xm, nullptr, NN_, DD, DD, ND, 0, ND, DD, DD, DD, 0, 0, 1, BB);

        // X = where(mask, Xm, X)
        updatex_kernel<<<gBND, TPB, 0, stream>>>(ix, Xm, X);
    }

    // logits = Xm[mask_idx] @ emb^T   -> [512, 32000]
    gather_kernel<<<(int)((((long)NMASK * DD) + TPB - 1) / TPB), TPB, 0, stream>>>(Xm, midx, Xgb);
    gemm(stream, Xgb, Eb, out, nullptr, NMASK, VOCAB_, DD, 0, 0, 0, DD, DD, VOCAB_, 0, 1, 0, 1);
}
